// UV_Aggregator_39075612459416
// MI455X (gfx1250) — matrix operand order repaired and checked by reading, not compiled, not run on hardware
//
#include <hip/hip_runtime.h>
#include <hip/hip_bf16.h>

// CDNA5 / gfx1250: wave32, WMMA 16x16x32 f16 -> f32 accumulate.
typedef __attribute__((ext_vector_type(16))) _Float16 v16h;
typedef __attribute__((ext_vector_type(8)))  float    v8f;

#define NNW 8          // waves (= nodes) per block
#define LHIST 50
#define DD 64

// ---- dynamic LDS layout (bytes) ----
// shared weights (per block)
#define OFF_W1T 0                      // Wt[n][k] f16 [64][128]
#define OFF_A1T (OFF_W1T + 64*128*2)   // lower half of A1, Wt layout [64][64]
#define OFF_W2T (OFF_A1T + 64*64*2)
#define OFF_A2T (OFF_W2T + 64*64*2)
#define OFF_B1  (OFF_A2T + 64*64*2)    // f32[64]
#define OFF_B2  (OFF_B1 + 256)
#define OFF_AB2 (OFF_B2 + 256)
#define OFF_A3  (OFF_AB2 + 256)
#define OFF_WAVE (OFF_A3 + 256)
// per-wave region
#define PW_BUF1 0                      // f16 [64][64]  o (persists for einsum)
#define PW_XT   8192                   // f16 [16][64]  per-tile scratch (x2 / h1)
#define PW_UVC  10240                  // f32[64]  ab1 + uv@A1[64:128]
#define PW_LOG  10496                  // f32[64]  attention logits / weights
#define PW_SIZE 10752
#define SMEM_BYTES (OFF_WAVE + NNW*PW_SIZE)

// A fragment (16x32 f16) from LDS, row-major halves, stride ld.
static __device__ inline v16h a_frag_lds(const _Float16* src, int ld, int row0,
                                         int k0, int lane) {
  const _Float16* p = src + (row0 + (lane & 15)) * ld + k0 + (lane >> 4) * 8;
  v16h a;
#pragma unroll
  for (int j = 0; j < 8; ++j) { a[j] = p[j]; a[8 + j] = p[16 + j]; }
  return a;
}

// B fragment (32x16 f16) from transposed weights Wt[n][k] (k contiguous, stride ldk).
static __device__ inline v16h b_frag(const _Float16* wt, int ldk, int k0,
                                     int n0, int lane) {
  const _Float16* p = wt + (n0 + (lane & 15)) * ldk + k0 + (lane >> 4) * 8;
  v16h b;
#pragma unroll
  for (int j = 0; j < 8; ++j) { b[j] = p[j]; b[8 + j] = p[16 + j]; }
  return b;
}

static __device__ inline v8f wmma16(v16h a, v16h b, v8f c) {
  return __builtin_amdgcn_wmma_f32_16x16x32_f16(false, a, false, b, (short)0, c,
                                                false, false);
}

// Store D tile with bias[col] + relu, converted to f16.
static __device__ inline void store_relu(_Float16* dst, int ld, int row0, int n0,
                                         int lane, v8f acc, const float* bias) {
  int col  = n0 + (lane & 15);
  int moff = (lane >> 4) * 8;
  float bv = bias[col];
#pragma unroll
  for (int r = 0; r < 8; ++r) {
    float v = acc[r] + bv;
    v = v > 0.f ? v : 0.f;
    dst[(row0 + r + moff) * ld + col] = (_Float16)v;
  }
}

__global__ __launch_bounds__(32 * NNW) void uv_agg_kernel(
    const int* __restrict__ nodes, const int* __restrict__ hist_uv,
    const int* __restrict__ hist_r, const int* __restrict__ hist_len,
    const float* __restrict__ v2e, const float* __restrict__ u2e,
    const float* __restrict__ r2e, const float* __restrict__ W1,
    const float* __restrict__ b1, const float* __restrict__ W2,
    const float* __restrict__ b2, const float* __restrict__ A1,
    const float* __restrict__ ab1, const float* __restrict__ A2,
    const float* __restrict__ ab2, const float* __restrict__ A3,
    const float* __restrict__ ab3, float* __restrict__ out, int nnodes) {
  extern __shared__ char smem[];
  _Float16* w1t = (_Float16*)(smem + OFF_W1T);
  _Float16* a1t = (_Float16*)(smem + OFF_A1T);
  _Float16* w2t = (_Float16*)(smem + OFF_W2T);
  _Float16* a2t = (_Float16*)(smem + OFF_A2T);
  float* sb1  = (float*)(smem + OFF_B1);
  float* sb2  = (float*)(smem + OFF_B2);
  float* sab2 = (float*)(smem + OFF_AB2);
  float* sa3  = (float*)(smem + OFF_A3);

  int tid = threadIdx.x;
  // ---- stage weights (transposed, f16) cooperatively ----
  for (int i = tid; i < 64 * 128; i += blockDim.x) {
    int n = i >> 7, k = i & 127;
    w1t[i] = (_Float16)W1[k * 64 + n];
  }
  for (int i = tid; i < 64 * 64; i += blockDim.x) {
    int n = i >> 6, k = i & 63;
    a1t[i] = (_Float16)A1[k * 64 + n];   // lower half of A1 only
    w2t[i] = (_Float16)W2[k * 64 + n];
    a2t[i] = (_Float16)A2[k * 64 + n];
  }
  if (tid < 64) {
    sb1[tid] = b1[tid]; sb2[tid] = b2[tid];
    sab2[tid] = ab2[tid]; sa3[tid] = A3[tid];
  }
  __syncthreads();

  int lane = tid & 31, wv = tid >> 5;
  int node = blockIdx.x * NNW + wv;
  if (node >= nnodes) return;

  char* pbase = smem + OFF_WAVE + wv * PW_SIZE;
  _Float16* buf1 = (_Float16*)(pbase + PW_BUF1);  // o
  _Float16* xt   = (_Float16*)(pbase + PW_XT);    // per-tile x2 / h1
  float* uvc     = (float*)(pbase + PW_UVC);
  float* logits  = (float*)(pbase + PW_LOG);

  // ---- self (user) embedding: 2 cols per lane, broadcast via shfl ----
  int uid = nodes[node];
  int c0 = lane * 2;
  float u0 = u2e[(size_t)uid * DD + c0];
  float u1 = u2e[(size_t)uid * DD + c0 + 1];

  // uvc[col] = ab1[col] + sum_k uv[k] * A1[64+k][col]  (row-constant attn bias)
  {
    float acc0 = ab1[c0], acc1 = ab1[c0 + 1];
#pragma unroll 8
    for (int k = 0; k < 64; ++k) {
      float u = __shfl((k & 1) ? u1 : u0, k >> 1);
      const float* row = A1 + (size_t)(64 + k) * DD;
      acc0 += u * row[c0];
      acc1 += u * row[c0 + 1];
    }
    uvc[c0] = acc0; uvc[c0 + 1] = acc1;
  }

  const int m = lane & 15, hs = lane >> 4;
  logits[lane] = 0.f;
  logits[lane + 32] = 0.f;

  // ================= fused per-tile pipeline =================
  // Per 16-row M-tile: gather -> GEMM1(W1) -> GEMM2(W2) -> GEMM3(A1) ->
  // GEMM4(A2)+A3-dot -> logits.  Everything row-block local.
  for (int mt = 0; mt < 4; ++mt) {
    int row = mt * 16 + m;

    // prefetch next tile's item-embedding row while this tile computes
    if (mt < 3) {
      int nrow = row + 16;
      if (nrow < LHIST) {
        int nuv = hist_uv[node * LHIST + nrow];
        __builtin_prefetch(v2e + (size_t)nuv * DD, 0, 0);
      }
    }

    // ---- gather A fragments for GEMM1 directly from global (f32 -> f16) ----
    v16h af[4];
    if (row < LHIST) {
      int uv = hist_uv[node * LHIST + row];
      int rr = hist_r[node * LHIST + row];
      const float* vrow = v2e + (size_t)uv * DD;
      const float* rrow = r2e + (size_t)rr * DD;
#pragma unroll
      for (int kt = 0; kt < 4; ++kt) {
        int khead = kt * 32 + hs * 8;
        v16h a;
        _Float16* ap = (_Float16*)&a;
#pragma unroll
        for (int c = 0; c < 2; ++c) {           // two 8-wide chunks: khead, khead+16
          int kc = khead + c * 16;
          const float* p = (kc < 64) ? (vrow + kc) : (rrow + kc - 64);
#pragma unroll
          for (int j = 0; j < 8; ++j) ap[c * 8 + j] = (_Float16)p[j];
        }
        af[kt] = a;
      }
    } else {
#pragma unroll
      for (int kt = 0; kt < 4; ++kt) af[kt] = (v16h)(_Float16)0.0f;  // zero pad
    }

    // ---- GEMM1: x2_tile = relu(X @ W1 + b1) -> xt ----
#pragma unroll
    for (int nt = 0; nt < 4; ++nt) {
      v8f acc = {};
#pragma unroll
      for (int kt = 0; kt < 4; ++kt)
        acc = wmma16(af[kt], b_frag(w1t, 128, kt * 32, nt * 16, lane), acc);
      store_relu(xt, DD, 0, nt * 16, lane, acc, sb1);
    }

    // ---- GEMM2: o_tile = relu(x2_tile @ W2 + b2) -> buf1 rows mt*16 ----
    {
      v16h a2[2];
#pragma unroll
      for (int kt = 0; kt < 2; ++kt) a2[kt] = a_frag_lds(xt, DD, 0, kt * 32, lane);
#pragma unroll
      for (int nt = 0; nt < 4; ++nt) {
        v8f acc = {};
#pragma unroll
        for (int kt = 0; kt < 2; ++kt)
          acc = wmma16(a2[kt], b_frag(w2t, DD, kt * 32, nt * 16, lane), acc);
        store_relu(buf1, DD, mt * 16, nt * 16, lane, acc, sb2);
      }
    }

    // ---- GEMM3: h1_tile = relu(o_tile @ A1[:64] + uvc) -> xt (reuse) ----
    {
      v16h a3[2];
#pragma unroll
      for (int kt = 0; kt < 2; ++kt)
        a3[kt] = a_frag_lds(buf1, DD, mt * 16, kt * 32, lane);
#pragma unroll
      for (int nt = 0; nt < 4; ++nt) {
        v8f acc = {};
#pragma unroll
        for (int kt = 0; kt < 2; ++kt)
          acc = wmma16(a3[kt], b_frag(a1t, DD, kt * 32, nt * 16, lane), acc);
        store_relu(xt, DD, 0, nt * 16, lane, acc, uvc);
      }
    }

    // ---- GEMM4 fused with A3 dot: logits[row] += relu(h1@A2+ab2)*A3[col] ----
    {
      v16h a4[2];
#pragma unroll
      for (int kt = 0; kt < 2; ++kt) a4[kt] = a_frag_lds(xt, DD, 0, kt * 32, lane);
#pragma unroll
      for (int nt = 0; nt < 4; ++nt) {
        v8f acc = {};
#pragma unroll
        for (int kt = 0; kt < 2; ++kt)
          acc = wmma16(a4[kt], b_frag(a2t, DD, kt * 32, nt * 16, lane), acc);
        int col = nt * 16 + (lane & 15);
        float a3v = sa3[col], ab2v = sab2[col];
        int moff = hs * 8;
#pragma unroll
        for (int r = 0; r < 8; ++r) {
          float v = acc[r] + ab2v;
          v = v > 0.f ? v : 0.f;
          atomicAdd(&logits[mt * 16 + r + moff], v * a3v);  // ds_add_f32
        }
      }
    }
  }

  // ---- masked softmax over history (ab3 is a uniform shift -> dropped) ----
  int hl = hist_len[node];
  float s1 = logits[lane], s2 = logits[lane + 32];
  bool v1 = lane < hl, v2 = (lane + 32) < hl;
  float mx = fmaxf(v1 ? s1 : -3.0e38f, v2 ? s2 : -3.0e38f);
#pragma unroll
  for (int o = 16; o; o >>= 1) mx = fmaxf(mx, __shfl_xor(mx, o));
  float e1 = v1 ? __expf(s1 - mx) : 0.f;
  float e2 = v2 ? __expf(s2 - mx) : 0.f;
  float sum = e1 + e2;
#pragma unroll
  for (int o = 16; o; o >>= 1) sum += __shfl_xor(sum, o);
  float inv = 1.f / sum;  // hl >= 1 so sum > 0
  logits[lane] = e1 * inv;
  logits[lane + 32] = e2 * inv;

  // ---- out[node] = att @ o  (2 cols per lane) ----
  float o0 = 0.f, o1 = 0.f;
#pragma unroll 5
  for (int r = 0; r < LHIST; ++r) {
    float a = logits[r];
    o0 += a * (float)buf1[r * DD + c0];
    o1 += a * (float)buf1[r * DD + c0 + 1];
  }
  out[(size_t)node * DD + c0] = o0;
  out[(size_t)node * DD + c0 + 1] = o1;
}

extern "C" void kernel_launch(void* const* d_in, const int* in_sizes, int n_in,
                              void* d_out, int out_size, void* d_ws, size_t ws_size,
                              hipStream_t stream) {
  (void)n_in; (void)out_size; (void)d_ws; (void)ws_size;
  const int*   nodes    = (const int*)d_in[0];
  const int*   hist_uv  = (const int*)d_in[1];
  const int*   hist_r   = (const int*)d_in[2];
  const int*   hist_len = (const int*)d_in[3];
  const float* v2e = (const float*)d_in[4];
  const float* u2e = (const float*)d_in[5];
  const float* r2e = (const float*)d_in[6];
  const float* W1  = (const float*)d_in[7];
  const float* b1  = (const float*)d_in[8];
  const float* W2  = (const float*)d_in[9];
  const float* b2  = (const float*)d_in[10];
  const float* A1  = (const float*)d_in[11];
  const float* ab1 = (const float*)d_in[12];
  const float* A2  = (const float*)d_in[13];
  const float* ab2 = (const float*)d_in[14];
  const float* A3  = (const float*)d_in[15];
  const float* ab3 = (const float*)d_in[16];
  float* out = (float*)d_out;

  int nnodes = in_sizes[0];
  dim3 block(32 * NNW);
  dim3 grid((nnodes + NNW - 1) / NNW);
  uv_agg_kernel<<<grid, block, SMEM_BYTES, stream>>>(
      nodes, hist_uv, hist_r, hist_len, v2e, u2e, r2e, W1, b1, W2, b2, A1, ab1,
      A2, ab2, A3, ab3, out, nnodes);
}
